// CAB_84533546320342
// MI455X (gfx1250) — matrix operand order repaired and checked by reading, not compiled, not run on hardware
//
#include <hip/hip_runtime.h>
#include <math.h>

// ---------------- problem constants ----------------
#define BB 8
#define CC 192
#define HH 128
#define WW 128
#define NN 16384            // H*W
#define HEADS 8
#define DH 24               // d_head
#define DP 32               // padded d_head
#define CP 256              // HEADS*DP padded channels
#define NT16 1024           // NN/16 spatial tiles of 16

typedef __attribute__((ext_vector_type(16))) __bf16 bf16x16;
typedef __attribute__((ext_vector_type(8)))  __bf16 bf16x8;
typedef __attribute__((ext_vector_type(8)))  float  f32x8;

__device__ inline f32x8 wmma_bf16(bf16x16 a, bf16x16 b, f32x8 c) {
  // D = A(16x32) * B(32x16) + C, f32 accumulate
  return __builtin_amdgcn_wmma_f32_16x16x32_bf16(false, a, false, b, (short)0, c,
                                                 false, false);
}
__device__ inline f32x8 zero8() {
  f32x8 z;
  #pragma unroll
  for (int i = 0; i < 8; ++i) z[i] = 0.f;
  return z;
}

// =====================================================================
// Pre-swizzle f32 weights (cout x cin) into bf16 B-fragment order:
// out[((ct*ksteps + ks)*32 + lane)*16 + e] = W[(ct*16 + lane%16)*cin
//                                             + ks*32 + (lane/16)*16 + e]
// -> in the GEMM each lane reads its whole fragment as 32 contiguous B.
// =====================================================================
__global__ __launch_bounds__(256) void swizzle_w(
    const float* __restrict__ W, __bf16* __restrict__ out, int cin) {
  const int idx = blockIdx.x * 256 + threadIdx.x;   // cout*cin total
  const int e  = idx & 15;
  const int L  = (idx >> 4) & 31;
  const int rest = idx >> 9;
  const int ksteps = cin >> 5;
  const int ks = rest % ksteps;
  const int ct = rest / ksteps;
  const int co = ct * 16 + (L & 15);
  const int ci = ks * 32 + (L >> 4) * 16 + e;
  out[idx] = (__bf16)W[(size_t)co * cin + ci];
}

// =====================================================================
// 1x1 conv as GEMM: out[b,co,n] = sum_ci W[co,ci] * in[b,ci,n]
// M = spatial(16), gemm-N = co (12 tiles = 192 per grid.y chunk), K = ci.
// Weights pre-swizzled bf16 (1 contiguous 32B load / fragment / lane).
// Input T = float (cvt in-register) or __bf16 (direct).
// =====================================================================
template <typename T>
__global__ __launch_bounds__(256) void conv1x1_wmma(
    const T* __restrict__ in, const __bf16* __restrict__ Ws,
    float* __restrict__ out, int cin, int cout_total) {
  const int lane = threadIdx.x & 31;
  const int wave = threadIdx.x >> 5;
  const int wid  = blockIdx.x * 8 + wave;          // [0, BB*NT16)
  const int b    = wid >> 10;
  const int n0   = (wid & (NT16 - 1)) << 4;
  const int m    = lane & 15;
  const int hi   = lane >> 4;
  const int col  = m;
  const int ct_base = blockIdx.y * 12;             // co chunk of 192
  const int ksteps = cin >> 5;

  const T* inb = in + (size_t)b * cin * (size_t)NN;

  f32x8 acc[12];
  #pragma unroll
  for (int t = 0; t < 12; ++t) acc[t] = zero8();

  for (int ks = 0; ks < ksteps; ++ks) {
    const int k0 = ks * 32;
    // A fragment: A[m][kk] = in[ci=k0+kk][n0+m]
    bf16x16 a;
    #pragma unroll
    for (int e = 0; e < 16; ++e) {
      int Ke = hi * 16 + e;
      a[e] = (__bf16)inb[(size_t)(k0 + Ke) * NN + n0 + m];
    }
    #pragma unroll
    for (int ct = 0; ct < 12; ++ct) {
      const bf16x16 bf = *(const bf16x16*)(
          Ws + ((size_t)((ct_base + ct) * ksteps + ks) * 32 + lane) * 16);
      acc[ct] = wmma_bf16(a, bf, acc[ct]);
    }
  }
  // D: lane holds 8 consecutive spatial values of column co
  #pragma unroll
  for (int ct = 0; ct < 12; ++ct) {
    int co = (ct_base + ct) * 16 + col;
    float* op = out + ((size_t)b * cout_total + co) * NN + n0 + hi * 8;
    float4 v0 = make_float4(acc[ct][0], acc[ct][1], acc[ct][2], acc[ct][3]);
    float4 v1 = make_float4(acc[ct][4], acc[ct][5], acc[ct][6], acc[ct][7]);
    *(float4*)op = v0;
    *((float4*)op + 1) = v1;
  }
}

// =====================================================================
// depthwise 3x3 (SAME), optional pos-embed add, + per-channel sum(x^2)
// =====================================================================
__global__ __launch_bounds__(256) void dw3x3_norm(
    const float* __restrict__ in, int in_cstride, int c_off,
    const float* __restrict__ w9, int w_off,
    const float* __restrict__ embed,
    float* __restrict__ out, float* __restrict__ sumsq) {
  const int tid = threadIdx.x;
  const size_t idx = (size_t)blockIdx.x * 256 + tid;
  const int n = (int)(idx % NN);
  const size_t bc = idx / NN;
  const int c = (int)(bc % CC);
  const int b = (int)(bc / CC);
  const int y = n >> 7, x = n & 127;
  const float* ip = in + ((size_t)b * in_cstride + c_off + c) * NN;
  const float* wp = w9 + (size_t)(w_off + c) * 9;
  float acc = 0.f;
  #pragma unroll
  for (int dy = -1; dy <= 1; ++dy)
    #pragma unroll
    for (int dx = -1; dx <= 1; ++dx) {
      int yy = y + dy, xx = x + dx;
      if (yy >= 0 && yy < HH && xx >= 0 && xx < WW)
        acc += wp[(dy + 1) * 3 + (dx + 1)] * ip[yy * WW + xx];
    }
  if (embed) acc += embed[c % DH];
  out[idx] = acc;
  __shared__ float red[256];
  red[tid] = acc * acc;
  __syncthreads();
  #pragma unroll
  for (int s = 128; s > 0; s >>= 1) {
    if (tid < s) red[tid] += red[tid + s];
    __syncthreads();
  }
  if (tid == 0) atomicAdd(&sumsq[b * CC + c], red[0]);
}

// depthwise 3x3 -> bf16 padded (b, CP, n) directly (v path, no norm)
__global__ __launch_bounds__(256) void dw3x3_v(
    const float* __restrict__ in, int in_cstride, int c_off,
    const float* __restrict__ w9, int w_off, __bf16* __restrict__ outp) {
  const size_t idx = (size_t)blockIdx.x * 256 + threadIdx.x;
  const int n = (int)(idx % NN);
  const size_t bc = idx / NN;
  const int c = (int)(bc % CC);
  const int b = (int)(bc / CC);
  const int y = n >> 7, x = n & 127;
  const float* ip = in + ((size_t)b * in_cstride + c_off + c) * NN;
  const float* wp = w9 + (size_t)(w_off + c) * 9;
  float acc = 0.f;
  #pragma unroll
  for (int dy = -1; dy <= 1; ++dy)
    #pragma unroll
    for (int dx = -1; dx <= 1; ++dx) {
      int yy = y + dy, xx = x + dx;
      if (yy >= 0 && yy < HH && xx >= 0 && xx < WW)
        acc += wp[(dy + 1) * 3 + (dx + 1)] * ip[yy * WW + xx];
    }
  int cp = (c / DH) * DP + (c % DH);
  outp[((size_t)b * CP + cp) * NN + n] = (__bf16)acc;
}

// scale by 1/max(||row||,eps), cast bf16, scatter to padded layout
__global__ __launch_bounds__(256) void scale_cast(
    const float* __restrict__ in, const float* __restrict__ sumsq,
    __bf16* __restrict__ outp) {
  const size_t idx = (size_t)blockIdx.x * 256 + threadIdx.x;
  const int n = (int)(idx % NN);
  const size_t bc = idx / NN;
  const int c = (int)(bc % CC);
  const int b = (int)(bc / CC);
  float s = sumsq[b * CC + c];
  float sc = 1.f / fmaxf(sqrtf(s), 1e-12f);
  int cp = (c / DH) * DP + (c % DH);
  outp[((size_t)b * CP + cp) * NN + n] = (__bf16)(in[idx] * sc);
}

// zero the pad channels (d = 24..31 of each head) of the 4 bf16 buffers
__global__ __launch_bounds__(256) void zero_pads(
    __bf16* q, __bf16* k, __bf16* v, __bf16* p) {
  const size_t idx = (size_t)blockIdx.x * 256 + threadIdx.x;   // B*HEADS*8*NN
  const int n = (int)(idx % NN);
  size_t r = idx / NN;
  const int dp = (int)(r % 8);
  const size_t bh = r / 8;
  const int h = (int)(bh % HEADS);
  const int b = (int)(bh / HEADS);
  size_t off = ((size_t)b * CP + h * DP + DH + dp) * NN + n;
  q[off] = (__bf16)0.f; k[off] = (__bf16)0.f;
  v[off] = (__bf16)0.f; p[off] = (__bf16)0.f;
}

// =====================================================================
// S[c,d] = sum_n Q[c,n] K[d,n]  per (b,h); 32x32 f32 out, K=16384 split 8
// ways across waves with f32 atomic reduction.
// =====================================================================
__global__ __launch_bounds__(256) void qk_wmma(
    const __bf16* __restrict__ Q, const __bf16* __restrict__ K,
    float* __restrict__ S) {
  const int lane = threadIdx.x & 31;
  const int wid  = blockIdx.x * 8 + (threadIdx.x >> 5);
  const int chunk = wid & 7;
  const int bh = wid >> 3;                 // [0, 64)
  const int m = lane & 15, hi = lane >> 4, col = m;

  const __bf16* Qb = Q + (size_t)bh * DP * NN;
  const __bf16* Kb = K + (size_t)bh * DP * NN;
  const __bf16* q0 = Qb + (size_t)m * NN;
  const __bf16* q1 = Qb + (size_t)(m + 16) * NN;
  const __bf16* k0r = Kb + (size_t)col * NN;
  const __bf16* k1r = Kb + (size_t)(col + 16) * NN;

  f32x8 acc[4];
  #pragma unroll
  for (int t = 0; t < 4; ++t) acc[t] = zero8();

  const int nbeg = chunk * 2048, nend = nbeg + 2048;
  for (int n0 = nbeg; n0 < nend; n0 += 32) {
    bf16x16 a0 = *(const bf16x16*)(q0 + n0 + hi * 16);
    bf16x16 a1 = *(const bf16x16*)(q1 + n0 + hi * 16);
    bf16x16 b0 = *(const bf16x16*)(k0r + n0 + hi * 16);
    bf16x16 b1 = *(const bf16x16*)(k1r + n0 + hi * 16);
    acc[0] = wmma_bf16(a0, b0, acc[0]);
    acc[1] = wmma_bf16(a0, b1, acc[1]);
    acc[2] = wmma_bf16(a1, b0, acc[2]);
    acc[3] = wmma_bf16(a1, b1, acc[3]);
  }
  float* Sb = S + (size_t)bh * 1024;
  #pragma unroll
  for (int t = 0; t < 4; ++t) {
    int c0 = (t >> 1) * 16, d0 = (t & 1) * 16;
    #pragma unroll
    for (int r = 0; r < 8; ++r)
      atomicAdd(&Sb[(c0 + hi * 8 + r) * 32 + d0 + col], acc[t][r]);
  }
}

// softmax over real 24 cols of real 24 rows; writes transposed bf16
// attnT[d*32+c], zero padding elsewhere. grid = B*HEADS, block = 32.
__global__ __launch_bounds__(32) void softmax32(
    const float* __restrict__ S, const float* __restrict__ temp,
    __bf16* __restrict__ attnT) {
  const int bh = blockIdx.x;
  const int c = threadIdx.x;
  const float t = temp[bh & (HEADS - 1)];
  __bf16* Ab = attnT + (size_t)bh * 1024;
  float vals[DH];
  #pragma unroll
  for (int d = 0; d < DH; ++d) vals[d] = 0.f;
  if (c < DH) {
    const float* Srow = S + (size_t)bh * 1024 + c * 32;
    float mx = -3.4e38f, sum = 0.f;
    #pragma unroll
    for (int d = 0; d < DH; ++d) { vals[d] = Srow[d] * t; mx = fmaxf(mx, vals[d]); }
    #pragma unroll
    for (int d = 0; d < DH; ++d) { vals[d] = __expf(vals[d] - mx); sum += vals[d]; }
    float inv = 1.f / sum;
    #pragma unroll
    for (int d = 0; d < DH; ++d) vals[d] *= inv;
  }
  #pragma unroll
  for (int d = 0; d < DP; ++d) {
    float o = (c < DH && d < DH) ? vals[d < DH ? d : 0] : 0.f;
    if (d >= DH) o = 0.f;
    Ab[d * 32 + c] = (__bf16)o;
  }
}

// =====================================================================
// PV + alpha blend: out[c,n] = sum_d attn[c,d] V[d,n]; both branches,
// U stored as bf16 (halves HBM traffic into the proj GEMM).
// =====================================================================
__global__ __launch_bounds__(256) void pv_wmma(
    const __bf16* __restrict__ V, const __bf16* __restrict__ A1T,
    const __bf16* __restrict__ A2T, const float* __restrict__ alpha,
    __bf16* __restrict__ U) {
  const int lane = threadIdx.x & 31;
  const int wid = blockIdx.x * 8 + (threadIdx.x >> 5);   // [0, 64*NT16)
  const int nt = wid & (NT16 - 1);
  const int bh = wid >> 10;
  const int h = bh & (HEADS - 1);
  const int b = bh >> 3;
  const int n0 = nt * 16;
  const int m = lane & 15, hi = lane >> 4, col = m;

  const __bf16* Vb = V + (size_t)bh * DP * NN;
  // A fragment from V: A[m][kk] = V[d=kk][n0+m]
  bf16x16 a;
  #pragma unroll
  for (int e = 0; e < 16; ++e) {
    int Ke = hi * 16 + e;
    a[e] = Vb[(size_t)Ke * NN + n0 + m];
  }
  f32x8 acc[2][2];
  #pragma unroll
  for (int br = 0; br < 2; ++br) {
    const __bf16* At = (br ? A2T : A1T) + (size_t)bh * 1024;
    #pragma unroll
    for (int ct = 0; ct < 2; ++ct) {
      bf16x16 bf;
      #pragma unroll
      for (int i = 0; i < 8; ++i) {
        bf[2 * i]     = At[(hi * 16 + 2 * i) * 32 + ct * 16 + col];
        bf[2 * i + 1] = At[(hi * 16 + 2 * i + 1) * 32 + ct * 16 + col];
      }
      acc[br][ct] = wmma_bf16(a, bf, zero8());
    }
  }
  #pragma unroll
  for (int ct = 0; ct < 2; ++ct) {
    int c = ct * 16 + col;
    float av = (c < DH) ? alpha[h * DH + c] : 0.f;
    bf16x8 st;
    #pragma unroll
    for (int e = 0; e < 8; ++e)
      st[e] = (__bf16)(av * acc[0][ct][e] + (1.f - av) * acc[1][ct][e]);
    __bf16* up = U + ((size_t)b * CP + h * DP + c) * NN + n0 + hi * 8;
    *(bf16x8*)up = st;
  }
}

// build zero-padded proj weight (192 x 256) f32
__global__ __launch_bounds__(256) void build_proj_pad(
    const float* __restrict__ pw, float* __restrict__ pp) {
  int idx = blockIdx.x * 256 + threadIdx.x;      // 192*256
  int co = idx / CP, cp = idx % CP;
  int h = cp >> 5, d = cp & 31;
  pp[idx] = (d < DH) ? pw[co * CC + h * DH + d] : 0.f;
}

// =====================================================================
extern "C" void kernel_launch(void* const* d_in, const int* in_sizes, int n_in,
                              void* d_out, int out_size, void* d_ws, size_t ws_size,
                              hipStream_t stream) {
  const float* x      = (const float*)d_in[0];
  const float* y      = (const float*)d_in[1];
  const float* q_w    = (const float*)d_in[2];
  const float* q_dw_w = (const float*)d_in[3];
  const float* kv_w   = (const float*)d_in[4];
  const float* kv_dw_w= (const float*)d_in[5];
  const float* pos_w  = (const float*)d_in[6];
  const float* proj_w = (const float*)d_in[7];
  const float* temp1  = (const float*)d_in[8];
  const float* temp2  = (const float*)d_in[9];
  const float* alpha  = (const float*)d_in[10];
  const float* pose   = (const float*)d_in[11];

  char* p = (char*)d_ws;
  auto alloc = [&](size_t bytes) -> void* {
    void* r = (void*)p; p += (bytes + 255) & ~(size_t)255; return r;
  };
  const size_t szCN = (size_t)BB * CC * NN * sizeof(float);       // 100.7 MB
  float*  qpre   = (float*)alloc(szCN);
  float*  kvpre  = (float*)alloc(2 * szCN);
  float*  qdw    = (float*)alloc(szCN);
  float*  kdw    = (float*)alloc(szCN);
  float*  posdw  = (float*)alloc(szCN);
  __bf16* qhat   = (__bf16*)alloc((size_t)BB * CP * NN * 2);
  __bf16* khat   = (__bf16*)alloc((size_t)BB * CP * NN * 2);
  __bf16* vhat   = (__bf16*)alloc((size_t)BB * CP * NN * 2);
  __bf16* poshat = (__bf16*)alloc((size_t)BB * CP * NN * 2);
  __bf16* Ubuf   = (__bf16*)alloc((size_t)BB * CP * NN * 2);
  float*  sumsq  = (float*)alloc(3 * BB * CC * 4);   // q | k | pos
  float*  S1     = (float*)alloc((size_t)BB * HEADS * 1024 * 4);
  float*  S2     = (float*)alloc((size_t)BB * HEADS * 1024 * 4);
  __bf16* At1    = (__bf16*)alloc((size_t)BB * HEADS * 1024 * 2);
  __bf16* At2    = (__bf16*)alloc((size_t)BB * HEADS * 1024 * 2);
  float*  ppad   = (float*)alloc((size_t)CC * CP * 4);
  __bf16* wsq    = (__bf16*)alloc((size_t)CC * CC * 2);       // swizzled q_w
  __bf16* wskv   = (__bf16*)alloc((size_t)2 * CC * CC * 2);   // swizzled kv_w
  __bf16* wsproj = (__bf16*)alloc((size_t)CC * CP * 2);       // swizzled padded proj
  float* sq_q = sumsq; float* sq_k = sumsq + BB * CC; float* sq_p = sumsq + 2 * BB * CC;

  const int grid_cn = (BB * CC * NN) / 256;          // elementwise grids

  // 0) weight prep: pad + swizzle into B-fragment order (bf16)
  swizzle_w<<<(CC * CC) / 256, 256, 0, stream>>>(q_w, wsq, CC);
  swizzle_w<<<(2 * CC * CC) / 256, 256, 0, stream>>>(kv_w, wskv, CC);
  build_proj_pad<<<(CC * CP) / 256, 256, 0, stream>>>(proj_w, ppad);
  swizzle_w<<<(CC * CP) / 256, 256, 0, stream>>>(ppad, wsproj, CP);

  // 1) pointwise convs (WMMA GEMMs)
  conv1x1_wmma<float><<<dim3(BB * NT16 / 8, 1), 256, 0, stream>>>(x, wsq, qpre, CC, CC);
  conv1x1_wmma<float><<<dim3(BB * NT16 / 8, 2), 256, 0, stream>>>(y, wskv, kvpre, CC, 2 * CC);

  // 2) zero reduction / score buffers
  (void)hipMemsetAsync(sumsq, 0, 3 * BB * CC * 4, stream);
  (void)hipMemsetAsync(S1, 0, (size_t)BB * HEADS * 1024 * 4, stream);
  (void)hipMemsetAsync(S2, 0, (size_t)BB * HEADS * 1024 * 4, stream);

  // 3) depthwise convs + sum-of-squares
  dw3x3_norm<<<grid_cn, 256, 0, stream>>>(qpre, CC, 0, q_dw_w, 0, nullptr, qdw, sq_q);
  dw3x3_norm<<<grid_cn, 256, 0, stream>>>(kvpre, 2 * CC, 0, kv_dw_w, 0, nullptr, kdw, sq_k);
  dw3x3_v   <<<grid_cn, 256, 0, stream>>>(kvpre, 2 * CC, CC, kv_dw_w, CC, vhat);
  dw3x3_norm<<<grid_cn, 256, 0, stream>>>(x, CC, 0, pos_w, 0, pose, posdw, sq_p);

  // 4) pad-channel zeroing + normalize/cast to padded bf16
  zero_pads<<<(BB * HEADS * 8 * NN) / 256, 256, 0, stream>>>(qhat, khat, vhat, poshat);
  scale_cast<<<grid_cn, 256, 0, stream>>>(qdw, sq_q, qhat);
  scale_cast<<<grid_cn, 256, 0, stream>>>(kdw, sq_k, khat);
  scale_cast<<<grid_cn, 256, 0, stream>>>(posdw, sq_p, poshat);

  // 5) channel-attention scores (K-split WMMA + atomics)
  qk_wmma<<<64, 256, 0, stream>>>(qhat, khat, S1);
  qk_wmma<<<64, 256, 0, stream>>>(poshat, khat, S2);

  // 6) softmax -> transposed bf16 attn
  softmax32<<<BB * HEADS, 32, 0, stream>>>(S1, temp1, At1);
  softmax32<<<BB * HEADS, 32, 0, stream>>>(S2, temp2, At2);

  // 7) attn @ V for both branches + alpha blend (WMMA, bf16 out)
  pv_wmma<<<(64 * NT16) / 8, 256, 0, stream>>>(vhat, At1, At2, alpha, Ubuf);

  // 8) projection conv (WMMA GEMM, bf16 input, padded 256 channels)
  conv1x1_wmma<__bf16><<<dim3(BB * NT16 / 8, 1), 256, 0, stream>>>(Ubuf, wsproj, (float*)d_out, CP, CC);
}
